// DeepBackflowNet_80229989089985
// MI455X (gfx1250) — matrix operand order repaired and checked by reading, not compiled, not run on hardware
//
#include <hip/hip_runtime.h>
#include <hip/hip_bf16.h>
#include <cmath>

typedef __attribute__((ext_vector_type(16))) _Float16 v16h;
typedef __attribute__((ext_vector_type(8)))  _Float16 v8h;
typedef __attribute__((ext_vector_type(8)))  float    v8f;

#define DEVINL __device__ __forceinline__

// ---- model dims ----
#define NWALK 128
#define NELEC 32
#define NNUCL 4
#define DMODEL 64
#define DINNER 128
#define DSTATE 16
#define SEQ    31
#define BROWS  4096      /* NWALK*NELEC            */
#define PROWS  131072    /* NWALK*NELEC*NELEC      */
#define MROWS  126976    /* BROWS*SEQ              */
#define MPAD   126992    /* padded rows for inp    */

// ============================================================================
// WMMA tile core: one wave computes a 16x16 f32 tile of Y = X @ W
//   A operand (16xK f32 -> f16): lane l (m=l&15,
//   hi=l>>4) reads contiguous K run [ks+16*hi, +16) per 32-K step (matches B).
//   B operand: weights pre-transposed+converted to f16 Wt[N][Kr]; lane l
//   (n = tile*16 + (l&15)) reads contiguous K run [ks+16*hi, +16).
// ============================================================================
template <class AF>
DEVINL v8f wmma_tile(AF af, const _Float16* __restrict__ Wt, int Kr, int ct) {
  const int l  = threadIdx.x & 31;
  const int m  = l & 15;
  const int hi = l >> 4;
  v8f c = {};
  const _Float16* bbase = Wt + (size_t)(ct * 16 + m) * Kr + hi * 16;
  for (int ks = 0; ks < Kr; ks += 32) {
    const float* p0 = af(m, ks + hi * 16);
    const float* p1 = af(m, ks + hi * 16 + 8);
    v8h alo, ahi;
#pragma unroll
    for (int i = 0; i < 8; ++i) { alo[i] = (_Float16)p0[i]; ahi[i] = (_Float16)p1[i]; }
    v16h a = __builtin_shufflevector(alo, ahi, 0,1,2,3,4,5,6,7,8,9,10,11,12,13,14,15);
    v16h b = *(const v16h*)(bbase + ks);
    c = __builtin_amdgcn_wmma_f32_16x16x32_f16(false, a, false, b, (short)0, c,
                                               false, false);
  }
  return c;
}

// ACT: 0=none 1=tanh 2=silu
template <int ACT>
DEVINL void store_tile(v8f c, float* __restrict__ Y, int ldy, int rowbase,
                       int colbase, const float* __restrict__ bias) {
  const int l  = threadIdx.x & 31;
  const int hi = l >> 4;
  const int n  = colbase + (l & 15);
  const float bv = bias[n];
#pragma unroll
  for (int vr = 0; vr < 8; ++vr) {
    int r = rowbase + vr + 8 * hi;
    float v = c[vr] + bv;
    if (ACT == 1) v = tanhf(v);
    if (ACT == 2) v = v / (1.f + expf(-v));
    Y[(size_t)r * ldy + n] = v;
  }
}

// ============================================================================
// Weight transpose + f32->f16 convert:  W[K][N] -> Wt[Nr][Kr], zero padded
// ============================================================================
__global__ __launch_bounds__(256)
void k_wt(const float* __restrict__ W, int K, int N, int Kr, int Nr,
          _Float16* __restrict__ Wt) {
  int e = blockIdx.x * 256 + threadIdx.x;
  if (e >= Nr * Kr) return;
  int n = e / Kr, k = e - n * Kr;
  float v = (n < N && k < K) ? W[(size_t)k * N + n] : 0.f;
  Wt[e] = (_Float16)v;
}

// ============================================================================
// Dense GEMM: 128 threads = 4 waves, wave = col tile, blockIdx.x = row tile.
// ============================================================================
template <int ACT>
__global__ __launch_bounds__(128)
void k_gemm_dense(const float* __restrict__ X, int ldx,
                  const _Float16* __restrict__ Wt, int Kr,
                  const float* __restrict__ bias,
                  float* __restrict__ Y, int ldy) {
  const int wave = threadIdx.x >> 5;
  const int ct = (int)blockIdx.y * 4 + wave;
  const int rowbase = (int)blockIdx.x * 16;
  __builtin_prefetch(X + (size_t)(rowbase + 16) * ldx, 0, 3);  // global_prefetch_b8
  auto af = [&](int mm, int k) -> const float* {
    return X + (size_t)(rowbase + mm) * ldx + k;
  };
  v8f c = wmma_tile(af, Wt, Kr, ct);
  store_tile<ACT>(c, Y, ldy, rowbase, ct * 16, bias);
}

// ============================================================================
// Gather+concat GEMM for proj: X row r=(b*31+j) is [h[w,idx], pair_h[w,e,idx]],
// K=128, fused into A-operand loads (8-float chunks never straddle col 64).
// Always SiLU.
// ============================================================================
__global__ __launch_bounds__(128)
void k_gemm_proj(const float* __restrict__ h, const float* __restrict__ ph,
                 const int* __restrict__ idx, const _Float16* __restrict__ Wt,
                 const float* __restrict__ bias, float* __restrict__ Y) {
  const int wave = threadIdx.x >> 5;
  const int rowbase = (int)blockIdx.x * 16;
  auto af = [&](int mm, int k) -> const float* {
    int r = rowbase + mm;
    int b = r / 31;
    int j = r - b * 31;
    int w = b >> 5, e = b & 31;
    int nb = idx[b * 31 + j];
    if (k < DMODEL) return h + (size_t)((w << 5) + nb) * DMODEL + k;
    return ph + (size_t)((((w << 5) + e) << 5) + nb) * DMODEL + (k - DMODEL);
  };
  v8f c = wmma_tile(af, Wt, 128, wave);
  store_tile<2>(c, Y, DMODEL, rowbase, wave * 16, bias);
}

// ============================================================================
// Features
// ============================================================================
__global__ __launch_bounds__(256)
void k_single_feats(const float* __restrict__ rE, const float* __restrict__ rN,
                    const float* __restrict__ chg, float* __restrict__ out) {
  int b = blockIdx.x * 256 + threadIdx.x;
  if (b >= BROWS) return;
  const float* r = rE + (size_t)b * 3;
  float* o = out + (size_t)b * 32;
  o[0] = r[0]; o[1] = r[1]; o[2] = r[2];
#pragma unroll
  for (int n = 0; n < NNUCL; ++n) {
    float dx = r[0] - rN[n * 3 + 0];
    float dy = r[1] - rN[n * 3 + 1];
    float dz = r[2] - rN[n * 3 + 2];
    float d = sqrtf(dx * dx + dy * dy + dz * dz);
    o[3 + n]  = d;
    o[7 + n]  = d * d;
    o[11 + n] = expf(-chg[n] * d);
  }
#pragma unroll
  for (int c = 15; c < 32; ++c) o[c] = 0.f;
}

__global__ __launch_bounds__(256)
void k_pair_feats(const float* __restrict__ rE, const int* __restrict__ spin,
                  float* __restrict__ pf, float* __restrict__ dist) {
  int p = blockIdx.x * 256 + threadIdx.x;
  if (p >= PROWS) return;
  int j = p & 31, i = (p >> 5) & 31, w = p >> 10;
  const float* ri = rE + (size_t)((w << 5) + i) * 3;
  const float* rj = rE + (size_t)((w << 5) + j) * 3;
  float vx = ri[0] - rj[0], vy = ri[1] - rj[1], vz = ri[2] - rj[2];
  float s = vx * vx + vy * vy + vz * vz;
  float dd = sqrtf(s > 0.f ? s : 1.f);
  float d = s > 0.f ? dd : 0.f;
  dist[p] = d;
  float* o = pf + (size_t)p * 32;
  o[0] = d; o[1] = d * d; o[2] = expf(-d);
  o[3] = vx; o[4] = vy; o[5] = vz;
  o[6] = (spin[i * 32 + j] != 0) ? 1.f : 0.f;
#pragma unroll
  for (int c = 7; c < 32; ++c) o[c] = 0.f;
}

// ============================================================================
// Stable 31-NN argsort per (w,e) row (insertion sort == jnp stable argsort)
// ============================================================================
__global__ __launch_bounds__(256)
void k_sort(const float* __restrict__ dist, int* __restrict__ idx) {
  int b = blockIdx.x * 256 + threadIdx.x;
  if (b >= BROWS) return;
  int e = b & 31;
  const float* dr = dist + (size_t)b * 32;
  float dv[32]; int iv[32];
  for (int j = 0; j < 32; ++j) {
    float v = (j == e) ? __builtin_inff() : dr[j];
    int k = j;
    while (k > 0 && dv[k - 1] > v) { dv[k] = dv[k - 1]; iv[k] = iv[k - 1]; --k; }
    dv[k] = v; iv[k] = j;
  }
  for (int j = 0; j < SEQ; ++j) idx[b * SEQ + j] = iv[j];
}

// ============================================================================
// Fused Mamba: one block (256 thr) per batch row b. Everything stays in LDS;
// WMMA for in_w (K=64,N=256) and x_w (K=128,N=36); VALU scan with state in
// VGPRs; mean-before-out_w since GEMM is linear.
// ============================================================================
__global__ __launch_bounds__(256)
void k_mamba(const float* __restrict__ inp, const _Float16* __restrict__ inWt,
             const float* __restrict__ in_b, const _Float16* __restrict__ xWt,
             const float* __restrict__ x_b, const float* __restrict__ dt_w,
             const float* __restrict__ dt_b, const float* __restrict__ A_log,
             const float* __restrict__ Dv, const float* __restrict__ out_w,
             const float* __restrict__ out_b, float* __restrict__ pagg) {
  __shared__ float s_x[32 * DINNER];     // xi, then xc in place
  __shared__ float s_z[32 * DINNER];     // silu(z)
  __shared__ float s_dt[32 * DINNER];
  __shared__ float s_xdbl[32 * 48];      // [dt_r(4) | B(16) | C(16)] padded
  __shared__ float s_part[2 * DINNER];
  __shared__ float s_g[DINNER];
  __shared__ float s_red[4 * DMODEL];

  const int tid = threadIdx.x;
  const int l = tid & 31, wave = tid >> 5, hi = l >> 4;
  const float* xbase = inp + (size_t)blockIdx.x * SEQ * DMODEL;

  // ---- stage 1: xz = inp @ in_w + in_b (M=32, N=256, K=64); split xi / silu(z)
  for (int it = 0; it < 4; ++it) {
    int tile = wave + it * 8;              // 0..31 = (rt:2)x(ct:16)
    int ct = tile & 15, rt = tile >> 4;
    auto af = [&](int mm, int k) -> const float* {
      return xbase + (size_t)(rt * 16 + mm) * DMODEL + k;
    };
    v8f c = wmma_tile(af, inWt, 64, ct);
    int n = ct * 16 + (l & 15);
    float bv = in_b[n];
#pragma unroll
    for (int vr = 0; vr < 8; ++vr) {
      int row = rt * 16 + vr + 8 * hi;
      float v = c[vr] + bv;
      if (n < DINNER) s_x[row * DINNER + n] = v;
      else            s_z[row * DINNER + (n - DINNER)] = v / (1.f + expf(-v));
    }
  }
  __syncthreads();

  // ---- stage 2: xc = silu((x[t-2]+x[t-1]+x[t])/3), in place (row 31 -> 0)
  {
    int t = tid >> 3, cb = (tid & 7) * 16;
    float r0[16], r1[16], r2[16];
#pragma unroll
    for (int i = 0; i < 16; ++i) {
      int d = cb + i;
      r0[i] = s_x[t * DINNER + d];
      r1[i] = (t >= 1) ? s_x[(t - 1) * DINNER + d] : 0.f;
      r2[i] = (t >= 2) ? s_x[(t - 2) * DINNER + d] : 0.f;
    }
    __syncthreads();
#pragma unroll
    for (int i = 0; i < 16; ++i) {
      float v = (r0[i] + r1[i] + r2[i]) * (1.f / 3.f);
      s_x[t * DINNER + cb + i] = (t < SEQ) ? v / (1.f + expf(-v)) : 0.f;
    }
  }
  __syncthreads();

  // ---- stage 3: xdbl = xc @ x_w + x_b  (M=32, N=36->48, K=128)
  if (wave < 6) {
    int rt = wave / 3, ct = wave - rt * 3;
    auto af = [&](int mm, int k) -> const float* {
      return &s_x[(rt * 16 + mm) * DINNER + k];
    };
    v8f c = wmma_tile(af, xWt, 128, ct);
    int n = ct * 16 + (l & 15);
    if (n < 36) {
      float bv = x_b[n];
#pragma unroll
      for (int vr = 0; vr < 8; ++vr)
        s_xdbl[(rt * 16 + vr + 8 * hi) * 48 + n] = c[vr] + bv;
    }
  }
  __syncthreads();

  // ---- stage 4: dt = softplus(xdbl[:, :4] @ dt_w + dt_b)
  {
    int t = tid >> 3, cb = (tid & 7) * 16;
#pragma unroll
    for (int i = 0; i < 16; ++i) {
      int d = cb + i;
      float acc = dt_b[d];
#pragma unroll
      for (int r = 0; r < 4; ++r) acc += s_xdbl[t * 48 + r] * dt_w[r * DINNER + d];
      s_dt[t * DINNER + d] = (acc > 20.f) ? acc : log1pf(expf(acc));
    }
  }
  __syncthreads();

  // ---- stage 5: selective scan; state in VGPRs (8 states per thread)
  {
    const int d = tid & 127, g = tid >> 7;
    float A8[8], hs[8];
#pragma unroll
    for (int i = 0; i < 8; ++i) {
      A8[i] = -expf(A_log[d * DSTATE + g * 8 + i]);
      hs[i] = 0.f;
    }
    const float Dd = Dv[d];
    float gacc = 0.f;
    for (int t = 0; t < SEQ; ++t) {
      float dtv = s_dt[t * DINNER + d];
      float xcv = s_x[t * DINNER + d];
      float part = 0.f;
#pragma unroll
      for (int i = 0; i < 8; ++i) {
        float Bt = s_xdbl[t * 48 + 4 + g * 8 + i];
        float Ct = s_xdbl[t * 48 + 20 + g * 8 + i];
        hs[i] = expf(A8[i] * dtv) * hs[i] + dtv * Bt * xcv;
        part += hs[i] * Ct;
      }
      s_part[g * DINNER + d] = part;
      __syncthreads();
      if (g == 0) {
        float y = s_part[d] + s_part[DINNER + d] + xcv * Dd;
        gacc += y * s_z[t * DINNER + d];
      }
      __syncthreads();
    }
    if (g == 0) s_g[d] = gacc * (1.f / 31.f);
  }
  __syncthreads();

  // ---- stage 6: pair_agg[b] = mean(y*silu(z)) @ out_w + out_b  (GEMV)
  {
    int n = tid & 63, q = tid >> 6;
    float acc = 0.f;
    for (int d = q * 32; d < q * 32 + 32; ++d) acc += s_g[d] * out_w[d * DMODEL + n];
    s_red[q * 64 + n] = acc;
    __syncthreads();
    if (tid < 64)
      pagg[(size_t)blockIdx.x * DMODEL + tid] =
          s_red[tid] + s_red[64 + tid] + s_red[128 + tid] + s_red[192 + tid] +
          out_b[tid];
  }
}

// ============================================================================
// LayerNorm over rows of 64: out = LN(a [+ b] [+ c]) * g + be; wave per row
// ============================================================================
__global__ __launch_bounds__(256)
void k_ln(const float* __restrict__ a, const float* __restrict__ b,
          const float* __restrict__ c, const float* __restrict__ g,
          const float* __restrict__ be, float* __restrict__ out, int rows) {
  int row = (int)blockIdx.x * 8 + (threadIdx.x >> 5);
  if (row >= rows) return;
  int l = threadIdx.x & 31;
  size_t o = (size_t)row * 64;
  float v0 = a[o + l]      + (b ? b[o + l]      : 0.f) + (c ? c[o + l]      : 0.f);
  float v1 = a[o + l + 32] + (b ? b[o + l + 32] : 0.f) + (c ? c[o + l + 32] : 0.f);
  float s = v0 + v1;
#pragma unroll
  for (int off = 16; off > 0; off >>= 1) s += __shfl_xor(s, off, 32);
  float mu = s * (1.f / 64.f);
  float d0 = v0 - mu, d1 = v1 - mu;
  float q = d0 * d0 + d1 * d1;
#pragma unroll
  for (int off = 16; off > 0; off >>= 1) q += __shfl_xor(q, off, 32);
  float inv = 1.f / sqrtf(q * (1.f / 64.f) + 1e-5f);
  out[o + l]      = d0 * inv * g[l]      + be[l];
  out[o + l + 32] = d1 * inv * g[l + 32] + be[l + 32];
}

// ============================================================================
// Host orchestration
// ============================================================================
extern "C" void kernel_launch(void* const* d_in, const int* in_sizes, int n_in,
                              void* d_out, int out_size, void* d_ws,
                              size_t ws_size, hipStream_t stream) {
  (void)in_sizes; (void)n_in; (void)out_size; (void)ws_size;
  const float* rE   = (const float*)d_in[0];
  const float* rN   = (const float*)d_in[1];
  const float* chg  = (const float*)d_in[2];
  const int*   spin = (const int*)  d_in[3];
  // params: top level insertion order, nested dict tree-flattened (sorted keys)
  // layer stride 24: 0:A_log 1:D 2:dt_b 3:dt_w 4:in_b 5:in_w 6:ln_p_b 7:ln_p_g
  // 8:ln_s_b 9:ln_s_g 10:out_b 11:out_w 12:pl_b1 13:pl_b2 14:pl_w1 15:pl_w2
  // 16:proj_b 17:proj_w 18:sl_b1 19:sl_b2 20:sl_w1 21:sl_w2 22:x_b 23:x_w
  auto LP = [&](int L, int o) -> const float* {
    return (const float*)d_in[4 + L * 24 + o];
  };
  const float* pe_b1 = (const float*)d_in[76];
  const float* pe_b2 = (const float*)d_in[77];
  const float* pe_w1 = (const float*)d_in[78];
  const float* pe_w2 = (const float*)d_in[79];
  const float* se_b1 = (const float*)d_in[80];
  const float* se_b2 = (const float*)d_in[81];
  const float* se_w1 = (const float*)d_in[82];
  const float* se_w2 = (const float*)d_in[83];

  char* wsp = (char*)d_ws;
  auto alloc = [&](size_t bytes) -> void* {
    void* p = (void*)wsp; wsp += (bytes + 255) & ~(size_t)255; return p;
  };
  _Float16 *inT[3], *xT[3], *projT[3], *sl1T[3], *sl2T[3], *pl1T[3], *pl2T[3];
  for (int L = 0; L < 3; ++L) {
    inT[L]   = (_Float16*)alloc(256 * 64 * 2);
    xT[L]    = (_Float16*)alloc(48 * 128 * 2);
    projT[L] = (_Float16*)alloc(64 * 128 * 2);
    sl1T[L]  = (_Float16*)alloc(64 * 64 * 2);
    sl2T[L]  = (_Float16*)alloc(64 * 64 * 2);
    pl1T[L]  = (_Float16*)alloc(64 * 64 * 2);
    pl2T[L]  = (_Float16*)alloc(64 * 64 * 2);
  }
  _Float16* seT1 = (_Float16*)alloc(64 * 32 * 2);
  _Float16* seT2 = (_Float16*)alloc(64 * 64 * 2);
  _Float16* peT1 = (_Float16*)alloc(64 * 32 * 2);
  _Float16* peT2 = (_Float16*)alloc(64 * 64 * 2);

  float* h     = (float*)alloc((size_t)BROWS * 64 * 4);
  float* ph    = (float*)alloc((size_t)PROWS * 64 * 4);
  float* dist  = (float*)alloc((size_t)PROWS * 4);
  float* sing  = (float*)alloc((size_t)BROWS * 32 * 4);
  float* pfeat = (float*)alloc((size_t)PROWS * 32 * 4);
  float* tbig  = (float*)alloc((size_t)PROWS * 64 * 4);
  float* ubig  = (float*)alloc((size_t)PROWS * 64 * 4);
  float* inpb  = (float*)alloc((size_t)MPAD * 64 * 4);
  float* ts1   = (float*)alloc((size_t)BROWS * 64 * 4);
  float* hself = (float*)alloc((size_t)BROWS * 64 * 4);
  float* pagg  = (float*)alloc((size_t)BROWS * 64 * 4);
  int*   idx   = (int*)  alloc((size_t)BROWS * SEQ * 4);

  auto wt = [&](const float* W, int K, int N, int Kr, int Nr, _Float16* dst) {
    int tot = Nr * Kr;
    k_wt<<<dim3((tot + 255) / 256), dim3(256), 0, stream>>>(W, K, N, Kr, Nr, dst);
  };
  for (int L = 0; L < 3; ++L) {
    wt(LP(L, 5), 64, 256, 64, 256, inT[L]);
    wt(LP(L, 23), 128, 36, 128, 48, xT[L]);
    wt(LP(L, 17), 128, 64, 128, 64, projT[L]);
    wt(LP(L, 20), 64, 64, 64, 64, sl1T[L]);
    wt(LP(L, 21), 64, 64, 64, 64, sl2T[L]);
    wt(LP(L, 14), 64, 64, 64, 64, pl1T[L]);
    wt(LP(L, 15), 64, 64, 64, 64, pl2T[L]);
  }
  wt(se_w1, 15, 64, 32, 64, seT1);
  wt(se_w2, 64, 64, 64, 64, seT2);
  wt(pe_w1, 7, 64, 32, 64, peT1);
  wt(pe_w2, 64, 64, 64, 64, peT2);

  // h = mlp2(single_input, se, tanh)
  k_single_feats<<<dim3(BROWS / 256), dim3(256), 0, stream>>>(rE, rN, chg, sing);
  k_gemm_dense<1><<<dim3(BROWS / 16, 1), dim3(128), 0, stream>>>(sing, 32, seT1, 32, se_b1, ts1, 64);
  k_gemm_dense<0><<<dim3(BROWS / 16, 1), dim3(128), 0, stream>>>(ts1, 64, seT2, 64, se_b2, h, 64);

  // pair_h = mlp2(pair_feats, pe, tanh)
  k_pair_feats<<<dim3(PROWS / 256), dim3(256), 0, stream>>>(rE, spin, pfeat, dist);
  k_gemm_dense<1><<<dim3(PROWS / 16, 1), dim3(128), 0, stream>>>(pfeat, 32, peT1, 32, pe_b1, tbig, 64);
  k_gemm_dense<0><<<dim3(PROWS / 16, 1), dim3(128), 0, stream>>>(tbig, 64, peT2, 64, pe_b2, ph, 64);

  // kNN indices (distance-independent of layers)
  k_sort<<<dim3(BROWS / 256), dim3(256), 0, stream>>>(dist, idx);

  for (int L = 0; L < 3; ++L) {
    // h_self = mlp2(h, sl, silu)
    k_gemm_dense<2><<<dim3(BROWS / 16, 1), dim3(128), 0, stream>>>(h, 64, sl1T[L], 64, LP(L, 18), ts1, 64);
    k_gemm_dense<0><<<dim3(BROWS / 16, 1), dim3(128), 0, stream>>>(ts1, 64, sl2T[L], 64, LP(L, 19), hself, 64);
    // inp = silu(concat(h_nb, p_nb) @ proj_w + proj_b)  (fused gather)
    k_gemm_proj<<<dim3(MROWS / 16), dim3(128), 0, stream>>>(h, ph, idx, projT[L], LP(L, 16), inpb);
    // pair_agg = mamba(inp).mean(seq)
    k_mamba<<<dim3(BROWS), dim3(256), 0, stream>>>(inpb, inT[L], LP(L, 4), xT[L], LP(L, 22),
                                                   LP(L, 3), LP(L, 2), LP(L, 0), LP(L, 1),
                                                   LP(L, 11), LP(L, 10), pagg);
    // pair_h = LN(pair_h + mlp2(pair_h, pl, silu))
    k_gemm_dense<2><<<dim3(PROWS / 16, 1), dim3(128), 0, stream>>>(ph, 64, pl1T[L], 64, LP(L, 12), tbig, 64);
    k_gemm_dense<0><<<dim3(PROWS / 16, 1), dim3(128), 0, stream>>>(tbig, 64, pl2T[L], 64, LP(L, 13), ubig, 64);
    k_ln<<<dim3(PROWS / 8), dim3(256), 0, stream>>>(ph, ubig, (const float*)nullptr,
                                                    LP(L, 7), LP(L, 6), ph, PROWS);
    // h = LN(h + h_self + pair_agg)
    float* hdst = (L == 2) ? (float*)d_out : h;
    k_ln<<<dim3(BROWS / 8), dim3(256), 0, stream>>>(h, hself, pagg,
                                                    LP(L, 9), LP(L, 8), hdst, BROWS);
  }
}
